// TaskConcept_39453569581465
// MI455X (gfx1250) — matrix operand order repaired and checked by reading, not compiled, not run on hardware
//
#include <hip/hip_runtime.h>

#define MARGIN 0.8f
#define USE_ASYNC_LDS 1

typedef __attribute__((ext_vector_type(16))) __bf16 v16bf;
typedef __attribute__((ext_vector_type(8)))  float  v8f;
typedef unsigned short ushort_t;

// ---------------------------------------------------------------------------
// CDNA5 async global->LDS (GLOBAL_LOAD_ASYNC_TO_LDS_B32, tracked by ASYNCcnt).
// LDS operand is the wave-relative byte offset = low 32 bits of the flat ptr
// (AS3->flat addrspacecast keeps the LDS offset in addr[31:0], ISA 10.2).
__device__ __forceinline__ void async_g2l_b32(void* lds, const void* g) {
  unsigned l = (unsigned)(unsigned long long)lds;
  unsigned long long ga = (unsigned long long)g;
  asm volatile("global_load_async_to_lds_b32 %0, %1, off"
               :: "v"(l), "v"(ga) : "memory");
}
__device__ __forceinline__ void wait_async0() {
  asm volatile("s_wait_asynccnt 0x0" ::: "memory");
}

// ---------------------------------------------------------------------------
__device__ __forceinline__ v8f wmma_bf16(v16bf a, v16bf b, v8f c) {
  return __builtin_amdgcn_wmma_f32_16x16x32_bf16(false, a, false, b, (short)0, c,
                                                 false, false);
}

// Packed-fragment load: 32 contiguous bytes per lane (2x global_load_b128).
__device__ __forceinline__ v16bf load_frag(const ushort_t* p, int tile) {
  const int lane = threadIdx.x & 31;
  return ((const v16bf*)p)[tile * 32 + lane];
}

// ---------------------------------------------------------------------------
// Pack f32 row-major A[M,K] into bf16 WMMA A-fragment tiles (16x32 per tile,
// tile index = mt*KT + kt, 32 lanes x 16 bf16 contiguous per tile).
// CDNA5 16-bit A layout: lane<16: M=lane, K in {0..7,16..23};
//                        lane>=16: M=lane-16, K in {8..15,24..31}.
__global__ void k_pack_a(const float* __restrict__ src, int M, int K,
                         ushort_t* __restrict__ dst) {
  const int KT = K >> 5;
  const int tiles = (M >> 4) * KT;
  const int tile = blockIdx.x * 8 + (threadIdx.x >> 5);
  if (tile >= tiles) return;
  const int mt = tile / KT, kt = tile % KT;
  const int lane = threadIdx.x & 31;
  const float* row = src + (mt * 16 + (lane & 15)) * K;
  const int kb = kt * 32 + ((lane & 16) ? 8 : 0);
  v16bf f;
#pragma unroll
  for (int e = 0; e < 16; ++e)
    f[e] = (__bf16)row[kb + ((e >> 3) << 4) + (e & 7)];
  ((v16bf*)dst)[tile * 32 + lane] = f;
}

// Pack f32 row-major B[K,N] into bf16 WMMA B-fragment tiles (32x16 per tile,
// tile index = nt*KT + kt).  Lane: N=lane&15; lanes<16 K in {0..7,16..23},
// lanes>=16 K in {8..15,24..31}.
__global__ void k_pack_b(const float* __restrict__ src, int K, int N,
                         ushort_t* __restrict__ dst) {
  const int KT = K >> 5, NT = N >> 4;
  const int tile = blockIdx.x * 8 + (threadIdx.x >> 5);
  if (tile >= KT * NT) return;
  const int nt = tile / KT, kt = tile % KT;
  const int lane = threadIdx.x & 31;
  const int col = nt * 16 + (lane & 15);
  const int kb = kt * 32 + ((lane & 16) ? 8 : 0);
  v16bf f;
#pragma unroll
  for (int e = 0; e < 16; ++e)
    f[e] = (__bf16)src[(kb + ((e >> 3) << 4) + (e & 7)) * N + col];
  ((v16bf*)dst)[tile * 32 + lane] = f;
}

// ---------------------------------------------------------------------------
// Triplet enumeration: labels[i] = i % 5 (5-way, 5-shot, N=25).
__device__ __forceinline__ int pos_idx(int a, int pi) {
  int la = a % 5, sa = a / 5;
  int sp = pi + (pi >= sa ? 1 : 0);
  return la + 5 * sp;
}
__device__ __forceinline__ int neg_idx(int a, int ni) {
  int la = a % 5;
  int c = ni & 3, sn = ni >> 2;
  int ln = c + (c >= la ? 1 : 0);
  return ln + 5 * sn;
}

// ---------------------------------------------------------------------------
// Kernel: per-task normalized pairwise L2 distances + semihard triplet count
__global__ void k_dist(const float* __restrict__ x, float* __restrict__ D,
                       float* __restrict__ cnt) {
  const int k = blockIdx.x;
  const float* X = x + k * 25 * 512;
  __shared__ float nrm[25];
  __shared__ float sD[625];
  __shared__ int scount;
  const int wave = threadIdx.x >> 5, lane = threadIdx.x & 31;

  if (threadIdx.x == 0) scount = 0;

  for (int i = wave; i < 25; i += 8) {
    float s = 0.f;
    for (int d = lane; d < 512; d += 32) { float v = X[i * 512 + d]; s += v * v; }
#pragma unroll
    for (int off = 16; off; off >>= 1) s += __shfl_down(s, off, 32);
    if (lane == 0) nrm[i] = sqrtf(s);
  }
  __syncthreads();

  for (int p = wave; p < 625; p += 8) {
    int i = p / 25, j = p % 25;
    float s = 0.f;
    for (int d = lane; d < 512; d += 32) s += X[i * 512 + d] * X[j * 512 + d];
#pragma unroll
    for (int off = 16; off; off >>= 1) s += __shfl_down(s, off, 32);
    if (lane == 0) {
      float cosij = s / (nrm[i] * nrm[j]);
      float dd = sqrtf(fmaxf(2.0f - 2.0f * cosij, 0.0f));
      sD[p] = dd;
      D[k * 625 + p] = dd;
    }
  }
  __syncthreads();

  int local = 0;
  for (int t = threadIdx.x; t < 2000; t += 256) {
    int a = t / 80, rem = t % 80;
    int p = pos_idx(a, rem / 20);
    int n = neg_idx(a, rem % 20);
    float tm = sD[a * 25 + n] - sD[a * 25 + p];
    if (tm > 0.0f && tm <= MARGIN) local++;
  }
  atomicAdd(&scount, local);
  __syncthreads();
  if (threadIdx.x == 0) cnt[k] = (float)scount;
}

// ---------------------------------------------------------------------------
// Kernel: APN[400,3072] = x_flat[400,512] @ {W1a|W1p|W1n}, packed bf16 WMMA.
// grid (6, 25), block 256 (8 waves); wave computes a 16x64 C strip.
// Ap: x packed (MT=25, KT=16). Bp: 3 packed blocks, each NT=64, KT=16.
__global__ void k_gemm1(const ushort_t* __restrict__ Ap,
                        const ushort_t* __restrict__ Bp,
                        float* __restrict__ APN) {
  const int wave = threadIdx.x >> 5, lane = threadIdx.x & 31;
  const int mt = blockIdx.y;
  const int nt0 = blockIdx.x * 32 + wave * 4;      // 6*32 = 192 n-tiles
  const int blk = nt0 >> 6;                        // groups of 4 never straddle
  const int ntL = nt0 & 63;
  const ushort_t* Bblk = Bp + (size_t)blk * (64 * 16 * 32 * 16);

  v8f c[4] = {};
  for (int kt = 0; kt < 16; ++kt) {
    v16bf a = load_frag(Ap, mt * 16 + kt);
#pragma unroll
    for (int t = 0; t < 4; ++t) {
      v16bf b = load_frag(Bblk, (ntL + t) * 16 + kt);
      c[t] = wmma_bf16(a, b, c[t]);
    }
  }
  const int col0 = lane & 15;
  const int r0 = (lane & 16) ? 8 : 0;
#pragma unroll
  for (int t = 0; t < 4; ++t) {
    int n = (nt0 + t) * 16;
#pragma unroll
    for (int r = 0; r < 8; ++r)
      APN[(mt * 16 + r0 + r) * 3072 + n + col0] = c[t][r];
  }
}

// ---------------------------------------------------------------------------
// Kernel: pooled_hidden[k,j] = sum over semihard triplets of
//         relu(A[k,a,j] + P[k,p,j] + N[k,n,j] + b1[j]).
// grid (8 j-chunks, 16 tasks), block 128; LDS staged via async loads.
__global__ void k_pool(const float* __restrict__ APN, const float* __restrict__ D,
                       const float* __restrict__ b1, float* __restrict__ PH) {
  const int k = blockIdx.y;
  const int j0 = blockIdx.x * 128;
  const int t = threadIdx.x;
  __shared__ float sA[25 * 128];
  __shared__ float sP[25 * 128];
  __shared__ float sN[25 * 128];
  __shared__ float sD[625];

  const int j = j0 + t;
  const float* base = APN + (k * 25) * 3072;
#if USE_ASYNC_LDS
  for (int i = 0; i < 25; ++i) {
    async_g2l_b32(&sA[i * 128 + t], base + i * 3072 + j);
    async_g2l_b32(&sP[i * 128 + t], base + i * 3072 + 1024 + j);
    async_g2l_b32(&sN[i * 128 + t], base + i * 3072 + 2048 + j);
  }
  for (int p = t; p < 625; p += 128)
    async_g2l_b32(&sD[p], D + k * 625 + p);
  wait_async0();
#else
  for (int i = 0; i < 25; ++i) {
    sA[i * 128 + t] = base[i * 3072 + j];
    sP[i * 128 + t] = base[i * 3072 + 1024 + j];
    sN[i * 128 + t] = base[i * 3072 + 2048 + j];
  }
  for (int p = t; p < 625; p += 128) sD[p] = D[k * 625 + p];
#endif
  __syncthreads();

  const float b1j = b1[j];
  float acc = 0.f;
  for (int a = 0; a < 25; ++a) {
    float Av = sA[a * 128 + t];
#pragma unroll
    for (int pi = 0; pi < 4; ++pi) {
      int p = pos_idx(a, pi);
      float dap = sD[a * 25 + p];
      float bap = Av + sP[p * 128 + t] + b1j;
#pragma unroll
      for (int ni = 0; ni < 20; ++ni) {
        int n = neg_idx(a, ni);
        float tm = sD[a * 25 + n] - dap;
        if (tm > 0.0f && tm <= MARGIN)
          acc += fmaxf(bap + sN[n * 128 + t], 0.0f);
      }
    }
  }
  PH[k * 1024 + j] = acc;
}

// ---------------------------------------------------------------------------
// Kernel: C[16,N] = A[16,K] @ B[K,N] + scale*bias (optional relu), packed ops.
// scale = cnt[row] if cnt != nullptr else 1.  One block, 8 waves, M=16 tile.
__global__ void k_gemm16(const ushort_t* __restrict__ Ap, int K,
                         const ushort_t* __restrict__ Bp, int N,
                         const float* __restrict__ bias,
                         const float* __restrict__ cnt, int doRelu,
                         float* __restrict__ C) {
  const int wave = threadIdx.x >> 5, lane = threadIdx.x & 31;
  const int KT = K >> 5;
  const int tilesPerWave = N >> 7;                 // (N/16)/8 waves
  for (int tt = 0; tt < tilesPerWave; ++tt) {
    const int nt = wave * tilesPerWave + tt;
    v8f c = {};
    for (int kt = 0; kt < KT; ++kt) {
      v16bf a = load_frag(Ap, kt);                 // MT = 1
      v16bf b = load_frag(Bp, nt * KT + kt);
      c = wmma_bf16(a, b, c);
    }
    const int col = nt * 16 + (lane & 15);
    const int r0 = (lane & 16) ? 8 : 0;
    const float bcol = bias[col];
#pragma unroll
    for (int r = 0; r < 8; ++r) {
      int row = r0 + r;
      float scale = cnt ? cnt[row] : 1.0f;
      float v = c[r] + scale * bcol;
      if (doRelu) v = fmaxf(v, 0.0f);
      C[row * N + col] = v;
    }
  }
}

// ---------------------------------------------------------------------------
// Kernel: score = O4[16,512] @ Wc[512,64] + bc, then row softmax -> out.
__global__ void k_tail4(const ushort_t* __restrict__ O4p,
                        const ushort_t* __restrict__ Wcp,
                        const float* __restrict__ bc, float* __restrict__ out) {
  __shared__ float S[16 * 64];
  const int wave = threadIdx.x >> 5, lane = threadIdx.x & 31;
  const int nt = wave;                             // 4 waves cover N=64
  v8f c = {};
  for (int kt = 0; kt < 16; ++kt) {
    v16bf a = load_frag(O4p, kt);
    v16bf b = load_frag(Wcp, nt * 16 + kt);
    c = wmma_bf16(a, b, c);
  }
  const int col = nt * 16 + (lane & 15);
  const int r0 = (lane & 16) ? 8 : 0;
#pragma unroll
  for (int r = 0; r < 8; ++r) S[(r0 + r) * 64 + col] = c[r] + bc[col];
  __syncthreads();

  if (threadIdx.x < 16) {
    int row = threadIdx.x;
    float mx = -1e30f;
    for (int jj = 0; jj < 64; ++jj) mx = fmaxf(mx, S[row * 64 + jj]);
    float sum = 0.f;
    for (int jj = 0; jj < 64; ++jj) sum += __expf(S[row * 64 + jj] - mx);
    float inv = 1.0f / sum;
    for (int jj = 0; jj < 64; ++jj)
      out[row * 64 + jj] = __expf(S[row * 64 + jj] - mx) * inv;
  }
}

// ---------------------------------------------------------------------------
extern "C" void kernel_launch(void* const* d_in, const int* in_sizes, int n_in,
                              void* d_out, int out_size, void* d_ws, size_t ws_size,
                              hipStream_t stream) {
  const float* x  = (const float*)d_in[0];   // [16,5,5,512]
  const float* W1 = (const float*)d_in[1];   // [1536,1024]
  const float* b1 = (const float*)d_in[2];   // [1024]
  const float* W2 = (const float*)d_in[3];   // [1024,512]
  const float* b2 = (const float*)d_in[4];   // [512]
  const float* W3 = (const float*)d_in[5];   // [512,1024]
  const float* b3 = (const float*)d_in[6];   // [1024]
  const float* W4 = (const float*)d_in[7];   // [1024,512]
  const float* b4 = (const float*)d_in[8];   // [512]
  const float* Wc = (const float*)d_in[9];   // [512,64]
  const float* bc = (const float*)d_in[10];  // [64]
  float* out = (float*)d_out;                // [16,64]
  (void)in_sizes; (void)n_in; (void)out_size; (void)ws_size;

  char* w = (char*)d_ws;
  auto alloc = [&](size_t bytes) {
    char* p = w;
    w += (bytes + 255) & ~(size_t)255;
    return p;
  };
  float* D    = (float*)alloc(16 * 625 * 4);
  float* cnt  = (float*)alloc(16 * 4);
  float* APN  = (float*)alloc((size_t)400 * 3072 * 4);
  float* PH   = (float*)alloc(16 * 1024 * 4);
  float* P2   = (float*)alloc(16 * 512 * 4);
  float* H3   = (float*)alloc(16 * 1024 * 4);
  float* O4   = (float*)alloc(16 * 512 * 4);
  ushort_t* xP  = (ushort_t*)alloc((size_t)400 * 512 * 2);
  ushort_t* W1p = (ushort_t*)alloc((size_t)1536 * 1024 * 2);
  ushort_t* W2p = (ushort_t*)alloc((size_t)1024 * 512 * 2);
  ushort_t* W3p = (ushort_t*)alloc((size_t)512 * 1024 * 2);
  ushort_t* W4p = (ushort_t*)alloc((size_t)1024 * 512 * 2);
  ushort_t* Wcp = (ushort_t*)alloc((size_t)512 * 64 * 2);
  ushort_t* PHp = (ushort_t*)alloc(16 * 1024 * 2);
  ushort_t* P2p = (ushort_t*)alloc(16 * 512 * 2);
  ushort_t* H3p = (ushort_t*)alloc(16 * 1024 * 2);
  ushort_t* O4p = (ushort_t*)alloc(16 * 512 * 2);

  auto packBlocks = [](int tiles) { return (tiles + 7) / 8; };

  k_dist<<<16, 256, 0, stream>>>(x, D, cnt);

  // one-shot operand packing (f32 -> bf16 fragment layout)
  k_pack_a<<<packBlocks(25 * 16), 256, 0, stream>>>(x, 400, 512, xP);
  for (int blk = 0; blk < 3; ++blk)
    k_pack_b<<<packBlocks(64 * 16), 256, 0, stream>>>(
        W1 + (size_t)blk * 512 * 1024, 512, 1024,
        W1p + (size_t)blk * 512 * 1024);
  k_pack_b<<<packBlocks(32 * 32), 256, 0, stream>>>(W2, 1024, 512, W2p);
  k_pack_b<<<packBlocks(64 * 16), 256, 0, stream>>>(W3, 512, 1024, W3p);
  k_pack_b<<<packBlocks(32 * 32), 256, 0, stream>>>(W4, 1024, 512, W4p);
  k_pack_b<<<packBlocks(4 * 16), 256, 0, stream>>>(Wc, 512, 64, Wcp);

  k_gemm1<<<dim3(6, 25), 256, 0, stream>>>(xP, W1p, APN);
  k_pool<<<dim3(8, 16), 128, 0, stream>>>(APN, D, b1, PH);

  k_pack_a<<<packBlocks(32), 256, 0, stream>>>(PH, 16, 1024, PHp);
  k_gemm16<<<1, 256, 0, stream>>>(PHp, 1024, W2p, 512, b2, cnt, 0, P2);
  k_pack_a<<<packBlocks(16), 256, 0, stream>>>(P2, 16, 512, P2p);
  k_gemm16<<<1, 256, 0, stream>>>(P2p, 512, W3p, 1024, b3, nullptr, 1, H3);
  k_pack_a<<<packBlocks(32), 256, 0, stream>>>(H3, 16, 1024, H3p);
  k_gemm16<<<1, 256, 0, stream>>>(H3p, 1024, W4p, 512, b4, nullptr, 0, O4);
  k_pack_a<<<packBlocks(16), 256, 0, stream>>>(O4, 16, 512, O4p);
  k_tail4<<<1, 128, 0, stream>>>(O4p, Wcp, bc, out);
}
